// GraphConvNet_27187142984247
// MI455X (gfx1250) — compile-verified, matrix operand order checked by reading, not run on hardware
//
#include <hip/hip_runtime.h>
#include <hip/hip_bf16.h>

// ---------------------------------------------------------------------------
// Types for CDNA5 WMMA bf16
// ---------------------------------------------------------------------------
typedef __bf16 bf16x16 __attribute__((ext_vector_type(16)));
typedef float  f32x8   __attribute__((ext_vector_type(8)));

union __align__(16) Frag16 {
    unsigned u[8];
    bf16x16  v;
};

__device__ __forceinline__ unsigned short f2bf(float f) {
    union { float f; unsigned u; } c; c.f = f;
    unsigned r = c.u + 0x7FFFu + ((c.u >> 16) & 1u);   // round-to-nearest-even
    return (unsigned short)(r >> 16);
}

// CDNA5 async global->LDS copy (16B per lane), tracked by ASYNCcnt.
__device__ __forceinline__ void async_b128(unsigned lds_off, const void* gaddr) {
    asm volatile("global_load_async_to_lds_b128 %0, %1, off"
                 :: "v"(lds_off), "v"(gaddr) : "memory");
}
__device__ __forceinline__ void wait_async0() {
    asm volatile("s_wait_asynccnt 0" ::: "memory");
}

// ---------------------------------------------------------------------------
// Tiled bf16 WMMA GEMM:  C[M,N] = sum_pairs A_p[M,K] * B_p^T
//   A : [M][K] row-major, lda == K
//   B : [N][K] row-major, ldb per pair (weights, xn, or pre-transposed acts)
// A and B tiles staged into LDS as [row][k] (stride 40 halfs) by
// GLOBAL_LOAD_ASYNC_TO_LDS_B128 (no VGPR staging, no spills); every fragment
// is then two 16B ds loads matching the documented 16-bit WMMA layout:
//   lane 0-15: row = lane,  VGPR v<4: K pair at half*8+2v ; v>=4: +16
// Double-buffered: next tile's async copies are issued right after the
// barrier and overlap with the 8 WMMAs on the current buffer.
// Epilogues:
//   0: outF = |c|                          (sim)
//   1: outB = bf16(relu(c + bias)) [+ outBT transposed copy]
//   2: outB = bf16(c / max(deg[m],1))      (mean aggregation)
//   3: outF = c + bias[n]                  (logits)
// ---------------------------------------------------------------------------
#define BM 128
#define BN 128
#define BK 32
#define LDS_S 40   // halfs per row: 80B, 16B-aligned, conflict-free

template<int EPI, int PAIRS>
__global__ __launch_bounds__(256) void gemm_bf16_wmma(
    const unsigned short* __restrict__ A0, const unsigned short* __restrict__ B0, int ldb0,
    const unsigned short* __restrict__ A1, const unsigned short* __restrict__ B1, int ldb1,
    int M, int N, int K,
    float* __restrict__ outF, unsigned short* __restrict__ outB,
    unsigned short* __restrict__ outBT, int ldT,
    const float* __restrict__ bias, const float* __restrict__ deg)
{
    __shared__ __align__(16) unsigned short sA[2][BM * LDS_S];
    __shared__ __align__(16) unsigned short sB[2][BN * LDS_S];

    const int tid   = threadIdx.x;
    const int lane  = tid & 31;
    const int wid   = tid >> 5;          // 0..7
    const int waveM = (wid >> 1) * 32;   // 0,32,64,96
    const int waveN = (wid & 1) * 64;    // 0,64
    const int blockM = blockIdx.y * BM;
    const int blockN = blockIdx.x * BN;

    // staging indices: 128 rows x 32 halfs = 512 x 16B; 4 async copies/thread
    const int r0 = tid >> 2;             // 0..63 (rows r0, r0+64)
    const int c8 = (tid & 3) * 8;        // 0,8,16,24

    const int half = lane >> 4;
    const int l15  = lane & 15;

    f32x8 acc[2][4] = {};

    for (int pair = 0; pair < PAIRS; ++pair) {
        const unsigned short* __restrict__ A = (pair == 0) ? A0 : A1;
        const unsigned short* __restrict__ B = (pair == 0) ? B0 : B1;
        const int ldb = (pair == 0) ? ldb0 : ldb1;
        const int nk  = K / BK;

        // issue async copies of one K tile into LDS buffer `buf`
        auto issue_tile = [&](int k0, int buf) {
            #pragma unroll
            for (int i = 0; i < 2; ++i) {
                int r = r0 + i * 64;
                unsigned offA = (unsigned)(size_t)&sA[buf][r * LDS_S + c8];
                async_b128(offA, A + (size_t)(blockM + r) * K + k0 + c8);
                int rbn = blockN + r;
                rbn = rbn < N ? rbn : N - 1;   // clamp: junk cols never written
                unsigned offB = (unsigned)(size_t)&sB[buf][r * LDS_S + c8];
                async_b128(offB, B + (size_t)rbn * ldb + k0 + c8);
            }
        };

        __syncthreads();          // previous pair's readers are done
        issue_tile(0, 0);
        int cur = 0;

        for (int kt = 0; kt < nk; ++kt) {
            wait_async0();        // own async writes for tile kt have landed
            __syncthreads();      // everyone's landed; other buffer is free
            const bool more = (kt + 1 < nk);
            if (more)             // overlap next tile's DMA with the WMMAs
                issue_tile((kt + 1) * BK, cur ^ 1);

            // ---- fragments (two 16B ds loads each) + 8 WMMAs
            Frag16 afr[2];
            #pragma unroll
            for (int s = 0; s < 2; ++s) {
                const unsigned short* base =
                    &sA[cur][(waveM + s * 16 + l15) * LDS_S + half * 8];
                ((uint4*)afr[s].u)[0] = *(const uint4*)base;
                ((uint4*)afr[s].u)[1] = *(const uint4*)(base + 16);
            }
            Frag16 bfr[4];
            #pragma unroll
            for (int s = 0; s < 4; ++s) {
                const unsigned short* base =
                    &sB[cur][(waveN + s * 16 + l15) * LDS_S + half * 8];
                ((uint4*)bfr[s].u)[0] = *(const uint4*)base;
                ((uint4*)bfr[s].u)[1] = *(const uint4*)(base + 16);
            }
            #pragma unroll
            for (int i = 0; i < 2; ++i)
                #pragma unroll
                for (int j = 0; j < 4; ++j)
                    acc[i][j] = __builtin_amdgcn_wmma_f32_16x16x32_bf16(
                        false, afr[i].v, false, bfr[j].v,
                        (short)0, acc[i][j], false, false);
            if (more) cur ^= 1;
        }
    }

    // ---- epilogue; C layout: n = lane&15, m = v + 8*(lane>>4)
    #pragma unroll
    for (int i = 0; i < 2; ++i) {
        #pragma unroll
        for (int j = 0; j < 4; ++j) {
            int baseM = blockM + waveM + i * 16 + half * 8;
            int baseN = blockN + waveN + j * 16 + l15;
            if (baseN >= N) continue;
            if constexpr (EPI == 0) {
                #pragma unroll
                for (int v = 0; v < 8; ++v)
                    outF[(size_t)(baseM + v) * N + baseN] = fabsf(acc[i][j][v]);
            } else if constexpr (EPI == 1) {
                __align__(16) unsigned short t[8];
                float b = bias ? bias[baseN] : 0.f;
                #pragma unroll
                for (int v = 0; v < 8; ++v) {
                    float c = acc[i][j][v] + b;
                    c = c > 0.f ? c : 0.f;
                    t[v] = f2bf(c);
                    outB[(size_t)(baseM + v) * N + baseN] = t[v];
                }
                if (outBT)   // transposed copy: 8 consecutive m -> one 16B store
                    *(uint4*)(outBT + (size_t)baseN * ldT + baseM) = *(uint4*)t;
            } else if constexpr (EPI == 2) {
                #pragma unroll
                for (int v = 0; v < 8; ++v) {
                    float d = deg[baseM + v];
                    d = d < 1.f ? 1.f : d;
                    outB[(size_t)(baseM + v) * N + baseN] = f2bf(acc[i][j][v] / d);
                }
            } else {
                #pragma unroll
                for (int v = 0; v < 8; ++v)
                    outF[(size_t)(baseM + v) * N + baseN] = acc[i][j][v] + bias[baseN];
            }
        }
    }
}

// ---------------------------------------------------------------------------
// Row L2-norm + fp32 -> bf16 converts (x and x/||x||); one wave per row
// ---------------------------------------------------------------------------
__global__ __launch_bounds__(256) void norm_convert_kernel(
    const float* __restrict__ x, unsigned short* __restrict__ xbf,
    unsigned short* __restrict__ xnbf, int n, int d)
{
    int row  = blockIdx.x * 8 + (threadIdx.x >> 5);
    int lane = threadIdx.x & 31;
    if (row >= n) return;
    const float* xr = x + (size_t)row * d;
    float ss = 0.f;
    for (int k = lane; k < d; k += 32) { float v = xr[k]; ss += v * v; }
    #pragma unroll
    for (int off = 16; off; off >>= 1) ss += __shfl_xor(ss, off, 32);
    float nrm = sqrtf(ss);
    nrm = nrm > 1e-8f ? nrm : 1e-8f;
    float inv = 1.f / nrm;
    for (int k = lane; k < d; k += 32) {
        float v = xr[k];
        xbf [(size_t)row * d + k] = f2bf(v);
        xnbf[(size_t)row * d + k] = f2bf(v * inv);
    }
}

__global__ __launch_bounds__(256) void f32_to_bf16_kernel(
    const float* __restrict__ s, unsigned short* __restrict__ d, int count)
{
    int i = blockIdx.x * 256 + threadIdx.x;
    if (i < count) d[i] = f2bf(s[i]);
}

// ---------------------------------------------------------------------------
// Per-row rank selection (quantile 'nearest'): 4-pass radix select on the
// fp32 bit patterns (sim >= 0, so uint order == float order). One block/row.
// ---------------------------------------------------------------------------
__global__ __launch_bounds__(256) void quantile_kernel(
    const float* __restrict__ sim, float* __restrict__ thr, int n, int rank)
{
    __shared__ unsigned hist[256];
    __shared__ unsigned s_sel[2];
    int row = blockIdx.x;
    const unsigned* srow = (const unsigned*)(sim + (size_t)row * n);
    unsigned prefix = 0, mask = 0, want = (unsigned)rank;
    for (int pass = 0; pass < 4; ++pass) {
        int shift = 24 - pass * 8;
        hist[threadIdx.x] = 0;
        __syncthreads();
        for (int i = threadIdx.x; i < n; i += 256) {
            unsigned v = srow[i];
            if ((v & mask) == prefix)
                atomicAdd(&hist[(v >> shift) & 0xFFu], 1u);
        }
        __syncthreads();
        if (threadIdx.x == 0) {
            unsigned cum = 0, b = 0;
            for (; b < 256; ++b) {
                unsigned c = hist[b];
                if (cum + c > want) break;
                cum += c;
            }
            s_sel[0] = b;
            s_sel[1] = want - cum;
        }
        __syncthreads();
        prefix |= (s_sel[0] << shift);
        mask   |= (0xFFu << shift);
        want    = s_sel[1];
        __syncthreads();
    }
    if (threadIdx.x == 0) {
        union { unsigned u; float f; } c; c.u = prefix;
        thr[row] = c.f;
    }
}

// ---------------------------------------------------------------------------
// Threshold + transpose: adjT[c][r] = (sim[r][c] >= thr[r]) ? bf16(1) : 0
// ---------------------------------------------------------------------------
__global__ __launch_bounds__(256) void threshold_kernel(
    const float* __restrict__ sim, const float* __restrict__ thr,
    unsigned short* __restrict__ adjT, int n)
{
    __shared__ unsigned short tile[64][65];
    int r0 = blockIdx.y * 64, c0 = blockIdx.x * 64;
    for (int idx = threadIdx.x; idx < 4096; idx += 256) {
        int i = idx >> 6, j = idx & 63;
        float v = sim[(size_t)(r0 + i) * n + c0 + j];
        tile[j][i] = (v >= thr[r0 + i]) ? (unsigned short)0x3F80u : (unsigned short)0;
    }
    __syncthreads();
    for (int idx = threadIdx.x; idx < 4096; idx += 256) {
        int c = idx >> 6, r = idx & 63;
        adjT[(size_t)(c0 + c) * n + r0 + r] = tile[c][r];
    }
}

// deg_in[c] = sum_r adjT[c][r]; one wave per row of adjT
__global__ __launch_bounds__(256) void deg_kernel(
    const unsigned short* __restrict__ adjT, float* __restrict__ deg, int n)
{
    int row  = blockIdx.x * 8 + (threadIdx.x >> 5);
    int lane = threadIdx.x & 31;
    if (row >= n) return;
    const unsigned short* r = adjT + (size_t)row * n;
    float s = 0.f;
    for (int k = lane; k < n; k += 32) s += (r[k] ? 1.f : 0.f);
    #pragma unroll
    for (int off = 16; off; off >>= 1) s += __shfl_xor(s, off, 32);
    if (lane == 0) deg[row] = s;
}

// Row softmax: 32 classes == 32 lanes, one wave per row
__global__ __launch_bounds__(256) void softmax_kernel(
    const float* __restrict__ logits, float* __restrict__ out, int n, int c)
{
    int row  = blockIdx.x * 8 + (threadIdx.x >> 5);
    int lane = threadIdx.x & 31;
    if (row >= n) return;
    float v = logits[(size_t)row * c + lane];
    float m = v;
    #pragma unroll
    for (int off = 16; off; off >>= 1) {
        float o = __shfl_xor(m, off, 32);
        m = m > o ? m : o;
    }
    float e = __expf(v - m);
    float s = e;
    #pragma unroll
    for (int off = 16; off; off >>= 1) s += __shfl_xor(s, off, 32);
    out[(size_t)row * c + lane] = e / s;
}

// ---------------------------------------------------------------------------
// Host launcher
// ---------------------------------------------------------------------------
extern "C" void kernel_launch(void* const* d_in, const int* in_sizes, int n_in,
                              void* d_out, int out_size, void* d_ws, size_t ws_size,
                              hipStream_t stream)
{
    const float* x     = (const float*)d_in[0];   // [8192,512]
    const float* w_in  = (const float*)d_in[1];   // [512,512]
    const float* w1_l  = (const float*)d_in[2];   // [512,512]
    const float* b1_l  = (const float*)d_in[3];   // [512]
    const float* w1_r  = (const float*)d_in[4];   // [512,512]
    const float* w2_l  = (const float*)d_in[5];   // [256,512]
    const float* b2_l  = (const float*)d_in[6];   // [256]
    const float* w2_r  = (const float*)d_in[7];   // [256,512]
    const float* w_out = (const float*)d_in[8];   // [32,256]
    const float* b_out = (const float*)d_in[9];   // [32]

    const int Nn = 8192, D = 512, H1 = 512, H2 = 256, C = 32;

    // ---- workspace carve-up
    char* p = (char*)d_ws;
    float*          sim   = (float*)p;          p += (size_t)Nn * Nn * 4;
    unsigned short* adjT  = (unsigned short*)p; p += (size_t)Nn * Nn * 2;
    unsigned short* xbf   = (unsigned short*)p; p += (size_t)Nn * D * 2;
    unsigned short* xnbf  = (unsigned short*)p; p += (size_t)Nn * D * 2;
    unsigned short* h0    = (unsigned short*)p; p += (size_t)Nn * H1 * 2;
    unsigned short* h0T   = (unsigned short*)p; p += (size_t)H1 * Nn * 2;
    unsigned short* aggb  = (unsigned short*)p; p += (size_t)Nn * H1 * 2;
    unsigned short* h1    = (unsigned short*)p; p += (size_t)Nn * H1 * 2;
    unsigned short* h1T   = (unsigned short*)p; p += (size_t)H1 * Nn * 2;
    unsigned short* h2    = (unsigned short*)p; p += (size_t)Nn * H2 * 2;
    float*          thr   = (float*)p;          p += (size_t)Nn * 4;
    float*          deg   = (float*)p;          p += (size_t)Nn * 4;
    float*          logit = (float*)p;          p += (size_t)Nn * C * 4;
    unsigned short* winb  = (unsigned short*)p; p += (size_t)D  * D * 2;
    unsigned short* w1lb  = (unsigned short*)p; p += (size_t)H1 * D * 2;
    unsigned short* w1rb  = (unsigned short*)p; p += (size_t)H1 * D * 2;
    unsigned short* w2lb  = (unsigned short*)p; p += (size_t)H2 * H1 * 2;
    unsigned short* w2rb  = (unsigned short*)p; p += (size_t)H2 * H1 * 2;
    unsigned short* wob   = (unsigned short*)p; p += (size_t)C  * H2 * 2;

    // ---- converts
    norm_convert_kernel<<<Nn / 8, 256, 0, stream>>>(x, xbf, xnbf, Nn, D);
    f32_to_bf16_kernel<<<(D * D + 255) / 256, 256, 0, stream>>>(w_in, winb, D * D);
    f32_to_bf16_kernel<<<(H1 * D + 255) / 256, 256, 0, stream>>>(w1_l, w1lb, H1 * D);
    f32_to_bf16_kernel<<<(H1 * D + 255) / 256, 256, 0, stream>>>(w1_r, w1rb, H1 * D);
    f32_to_bf16_kernel<<<(H2 * H1 + 255) / 256, 256, 0, stream>>>(w2_l, w2lb, H2 * H1);
    f32_to_bf16_kernel<<<(H2 * H1 + 255) / 256, 256, 0, stream>>>(w2_r, w2rb, H2 * H1);
    f32_to_bf16_kernel<<<(C * H2 + 255) / 256, 256, 0, stream>>>(w_out, wob, C * H2);

    // ---- sim = |xn @ xn^T|   (B == xn, stored [N][K])
    gemm_bf16_wmma<0, 1><<<dim3(Nn / BN, Nn / BM), 256, 0, stream>>>(
        xnbf, xnbf, D, nullptr, nullptr, 0, Nn, Nn, D,
        sim, nullptr, nullptr, 0, nullptr, nullptr);

    // ---- per-row quantile (nearest rank of 0.9*(n-1) = 7372)
    {
        int rank = (int)(0.9f * (float)(Nn - 1) + 0.5f);
        quantile_kernel<<<Nn, 256, 0, stream>>>(sim, thr, Nn, rank);
    }

    // ---- adjT (bf16 0/1, transposed) + in-degrees
    threshold_kernel<<<dim3(Nn / 64, Nn / 64), 256, 0, stream>>>(sim, thr, adjT, Nn);
    deg_kernel<<<Nn / 8, 256, 0, stream>>>(adjT, deg, Nn);

    // ---- h0 = relu(x @ w_in^T)  (also emit h0T for the aggregation GEMM)
    gemm_bf16_wmma<1, 1><<<dim3(H1 / BN, Nn / BM), 256, 0, stream>>>(
        xbf, winb, D, nullptr, nullptr, 0, Nn, H1, D,
        nullptr, h0, h0T, Nn, nullptr, nullptr);

    // ---- layer 1: agg = (adjT @ h0)/deg ; h1 = relu(agg@w1_l^T + h0@w1_r^T + b1)
    gemm_bf16_wmma<2, 1><<<dim3(H1 / BN, Nn / BM), 256, 0, stream>>>(
        adjT, h0T, Nn, nullptr, nullptr, 0, Nn, H1, Nn,
        nullptr, aggb, nullptr, 0, nullptr, deg);
    gemm_bf16_wmma<1, 2><<<dim3(H1 / BN, Nn / BM), 256, 0, stream>>>(
        aggb, w1lb, H1, h0, w1rb, H1, Nn, H1, H1,
        nullptr, h1, h1T, Nn, b1_l, nullptr);

    // ---- layer 2: agg = (adjT @ h1)/deg ; h2 = relu(agg@w2_l^T + h1@w2_r^T + b2)
    gemm_bf16_wmma<2, 1><<<dim3(H1 / BN, Nn / BM), 256, 0, stream>>>(
        adjT, h1T, Nn, nullptr, nullptr, 0, Nn, H1, Nn,
        nullptr, aggb, nullptr, 0, nullptr, deg);
    gemm_bf16_wmma<1, 2><<<dim3((H2 + BN - 1) / BN, Nn / BM), 256, 0, stream>>>(
        aggb, w2lb, H1, h1, w2rb, H1, Nn, H2, H1,
        nullptr, h2, nullptr, 0, b2_l, nullptr);

    // ---- logits = h2 @ w_out^T + b_out ; softmax -> d_out
    gemm_bf16_wmma<3, 1><<<dim3((C + BN - 1) / BN, Nn / BM), 256, 0, stream>>>(
        h2, wob, H2, nullptr, nullptr, 0, Nn, C, H2,
        logit, nullptr, nullptr, 0, b_out, nullptr);
    softmax_kernel<<<Nn / 8, 256, 0, stream>>>(logit, (float*)d_out, Nn, C);
}
